// DecodePIFLayer_74921409511747
// MI455X (gfx1250) — compile-verified, matrix operand order checked by reading, not run on hardware
//
#include <hip/hip_runtime.h>

// Problem constants (from reference: B=4, h=68, w=120, STRIDE=8)
#define BB      4
#define HF      544            // 68*8
#define WF      960            // 120*8
#define KJ      8160           // 68*120 cells
#define KSTEP   32             // K of v_wmma_f32_16x16x32_f16
#define NKSTEP  255            // 8160 / 32 (exact)
#define MB_     17             // 544 / 32  (M blocks per batch, 32 rows per wave)
#define NB_     15             // 960 / 64  (N blocks per batch, 64 cols per wave)
#define MIN_CONF 0.1f

typedef __attribute__((ext_vector_type(16))) _Float16 v16h;
typedef __attribute__((ext_vector_type(8)))  _Float16 v8h;
typedef __attribute__((ext_vector_type(8)))  float    v8f;

// ---------------------------------------------------------------------------
// Kernel 1a: gyT[b][y][j] = exp(-(y - my[b,j])^2 / (2 var[b,j]))   (f16, K-major)
// ---------------------------------------------------------------------------
__global__ __launch_bounds__(256) void pif_gen_gy(const float* __restrict__ mean,
                                                  const float* __restrict__ var,
                                                  _Float16* __restrict__ gyT) {
    size_t idx = (size_t)blockIdx.x * 256 + threadIdx.x;
    const size_t total = (size_t)BB * HF * KJ;
    if (idx >= total) return;
    int j = (int)(idx % KJ);
    int y = (int)((idx / KJ) % HF);
    int b = (int)(idx / ((size_t)KJ * HF));
    float my    = mean[((size_t)(b * KJ + j)) * 2 + 1];
    float inv2v = 0.5f / var[b * KJ + j];
    float d     = (float)y - my;
    gyT[idx] = (_Float16)__expf(-d * d * inv2v);
}

// ---------------------------------------------------------------------------
// Kernel 1b: gxT[b][x][j] = exp(-(x - mx[b,j])^2 / (2 var[b,j])) * conf_masked
// ---------------------------------------------------------------------------
__global__ __launch_bounds__(256) void pif_gen_gx(const float* __restrict__ mean,
                                                  const float* __restrict__ var,
                                                  const float* __restrict__ conf,
                                                  _Float16* __restrict__ gxT) {
    size_t idx = (size_t)blockIdx.x * 256 + threadIdx.x;
    const size_t total = (size_t)BB * WF * KJ;
    if (idx >= total) return;
    int j = (int)(idx % KJ);
    int x = (int)((idx / KJ) % WF);
    int b = (int)(idx / ((size_t)KJ * WF));
    float mx    = mean[((size_t)(b * KJ + j)) * 2 + 0];
    float inv2v = 0.5f / var[b * KJ + j];
    float c     = conf[b * KJ + j];
    float cm    = (c > MIN_CONF) ? c : 0.0f;
    float d     = (float)x - mx;
    gxT[idx] = (_Float16)(__expf(-d * d * inv2v) * cm);
}

// ---------------------------------------------------------------------------
// Kernel 2: batched GEMM  out[b] = gyT[b] (544xK) * gxT[b]^T (Kx960)
// via v_wmma_f32_16x16x32_f16. One wave computes a 32x64 output block
// (2 M-tiles x 4 N-tiles). K-major storage makes each lane's fragment two
// contiguous 16-byte chunks per the CDNA5 16-bit A-matrix VGPR layout:
//   lanes 0-15 : K = k0+{0..7}  then k0+{16..23}
//   lanes 16-31: K = k0+{8..15} then k0+{24..31}
//
// Job decomposition: mB varies fastest, so the 4 waves of a workgroup
// (= 4 SIMDs of one WGP) read the SAME B tile -> WGP$ hits cut L2 traffic
// on the dominant operand ~4x.
//
// Register double-buffering (ping-pong fragment sets, unroll-by-2) issues
// each k-step's 12 b128 loads one full iteration ahead of their WMMAs.
// ---------------------------------------------------------------------------
__global__ __launch_bounds__(128) void pif_wmma_gemm(const _Float16* __restrict__ gyT,
                                                     const _Float16* __restrict__ gxT,
                                                     float* __restrict__ out) {
    const int wave = blockIdx.x * 4 + (threadIdx.x >> 5);
    const int lane = threadIdx.x & 31;

    const int per_b = MB_ * NB_;          // 255 wave-jobs per batch
    const int b   = wave / per_b;
    const int rem = wave % per_b;
    const int m0  = (rem % MB_) * 32;     // mB fastest: waves in a WGP share B
    const int n0  = (rem / MB_) * 64;

    const int laneM = lane & 15;
    const int koff  = (lane >> 4) * 8;    // lanes 16-31 start at K+8

    const _Float16* pa0 = gyT + ((size_t)(b * HF + m0 + laneM)) * KJ + koff;
    const _Float16* pa1 = pa0 + (size_t)16 * KJ;
    const _Float16* pb0 = gxT + ((size_t)(b * WF + n0 + laneM)) * KJ + koff;
    const _Float16* pb1 = pb0 + (size_t)16 * KJ;
    const _Float16* pb2 = pb0 + (size_t)32 * KJ;
    const _Float16* pb3 = pb0 + (size_t)48 * KJ;

    v8f acc[2][4] = {};

#define LOAD_FRAG(P, KB)                                                       \
    __builtin_shufflevector(*(const v8h*)((P) + (KB)),                         \
                            *(const v8h*)((P) + (KB) + 16),                    \
                            0,1,2,3,4,5,6,7,8,9,10,11,12,13,14,15)

#define LOAD_SET(SUF, KB)                                                      \
    A0##SUF = LOAD_FRAG(pa0, (KB)); A1##SUF = LOAD_FRAG(pa1, (KB));            \
    B0##SUF = LOAD_FRAG(pb0, (KB)); B1##SUF = LOAD_FRAG(pb1, (KB));            \
    B2##SUF = LOAD_FRAG(pb2, (KB)); B3##SUF = LOAD_FRAG(pb3, (KB))

#define WMMA_SET(SUF)                                                                                          \
    acc[0][0] = __builtin_amdgcn_wmma_f32_16x16x32_f16(false, A0##SUF, false, B0##SUF, (short)0, acc[0][0], false, false); \
    acc[0][1] = __builtin_amdgcn_wmma_f32_16x16x32_f16(false, A0##SUF, false, B1##SUF, (short)0, acc[0][1], false, false); \
    acc[0][2] = __builtin_amdgcn_wmma_f32_16x16x32_f16(false, A0##SUF, false, B2##SUF, (short)0, acc[0][2], false, false); \
    acc[0][3] = __builtin_amdgcn_wmma_f32_16x16x32_f16(false, A0##SUF, false, B3##SUF, (short)0, acc[0][3], false, false); \
    acc[1][0] = __builtin_amdgcn_wmma_f32_16x16x32_f16(false, A1##SUF, false, B0##SUF, (short)0, acc[1][0], false, false); \
    acc[1][1] = __builtin_amdgcn_wmma_f32_16x16x32_f16(false, A1##SUF, false, B1##SUF, (short)0, acc[1][1], false, false); \
    acc[1][2] = __builtin_amdgcn_wmma_f32_16x16x32_f16(false, A1##SUF, false, B2##SUF, (short)0, acc[1][2], false, false); \
    acc[1][3] = __builtin_amdgcn_wmma_f32_16x16x32_f16(false, A1##SUF, false, B3##SUF, (short)0, acc[1][3], false, false)

    v16h A0c, A1c, B0c, B1c, B2c, B3c;    // current fragment set
    v16h A0n, A1n, B0n, B1n, B2n, B3n;    // next fragment set

    LOAD_SET(c, 0);                       // preload k-step 0

    // NKSTEP = 255 (odd): pairs (0,1)..(252,253), tail k-step 254 after loop.
    for (int kk = 0; kk + 1 < NKSTEP; kk += 2) {
        const size_t kb1 = (size_t)(kk + 1) * KSTEP;
        const size_t kb2 = (size_t)(kk + 2) * KSTEP;   // kk+2 <= 254 always valid
        LOAD_SET(n, kb1);                 // loads land while WMMAs below run
        WMMA_SET(c);
        LOAD_SET(c, kb2);
        WMMA_SET(n);
    }
    WMMA_SET(c);                          // final k-step 254

#undef WMMA_SET
#undef LOAD_SET
#undef LOAD_FRAG

    // Epilogue: 32-bit C/D layout — VGPR i: lanes 0-15 -> M=i, lanes 16-31 -> M=i+8
    const int rowOff = (lane >> 4) * 8;
    const int colN   = lane & 15;
#pragma unroll
    for (int t = 0; t < 2; ++t) {
#pragma unroll
        for (int u = 0; u < 4; ++u) {
            float* basep = out + ((size_t)(b * HF + m0 + 16 * t + rowOff)) * WF
                               + n0 + 16 * u + colN;
#pragma unroll
            for (int i = 0; i < 8; ++i)
                basep[(size_t)i * WF] = acc[t][u][i];
        }
    }
}

// ---------------------------------------------------------------------------
extern "C" void kernel_launch(void* const* d_in, const int* in_sizes, int n_in,
                              void* d_out, int out_size, void* d_ws, size_t ws_size,
                              hipStream_t stream) {
    const float* mean       = (const float*)d_in[0];   // [B,h,w,2]
    const float* variance   = (const float*)d_in[1];   // [B,h,w]
    const float* confidence = (const float*)d_in[2];   // [B,h,w]
    float* out = (float*)d_out;                        // [B,544,960]

    _Float16* gyT = (_Float16*)d_ws;                           // [B][544][K] f16
    _Float16* gxT = gyT + (size_t)BB * HF * KJ;                // [B][960][K] f16

    // 1) materialize separable Gaussians in f16, K-major
    {
        size_t totY = (size_t)BB * HF * KJ;
        size_t totX = (size_t)BB * WF * KJ;
        dim3 gY((unsigned)((totY + 255) / 256));
        dim3 gX((unsigned)((totX + 255) / 256));
        pif_gen_gy<<<gY, 256, 0, stream>>>(mean, variance, gyT);
        pif_gen_gx<<<gX, 256, 0, stream>>>(mean, variance, confidence, gxT);
    }

    // 2) batched WMMA GEMM: 1020 waves (4 per 128-thread block) -> 255 blocks
    pif_wmma_gemm<<<MB_ * NB_ * BB / 4, 128, 0, stream>>>(gyT, gxT, out);
}
